// NanoporeVQModel_21182778703879
// MI455X (gfx1250) — matrix operand order repaired and checked by reading, not compiled, not run on hardware
//
#include <hip/hip_runtime.h>
#include <cstdint>
#include <cstddef>

// ---------------- CDNA5 WMMA plumbing ----------------
typedef __attribute__((ext_vector_type(16))) __bf16 bf16x16;
typedef __attribute__((ext_vector_type(8)))  float  f32x8;
typedef unsigned short bfr;              // raw bf16 storage in memory

union FragBF { bf16x16 v; unsigned int u[8]; };

__device__ __forceinline__ unsigned short f2bf(float f) {
  unsigned u = __builtin_bit_cast(unsigned, f);
  unsigned r = (u + 0x7FFFu + ((u >> 16) & 1u)) >> 16;   // round-to-nearest-even
  return (unsigned short)r;
}
__device__ __forceinline__ float bf2f(unsigned short h) {
  return __builtin_bit_cast(float, ((unsigned)h) << 16);
}
__device__ __forceinline__ unsigned int pack_bf2(float a, float b) {
  return (unsigned)f2bf(a) | ((unsigned)f2bf(b) << 16);
}
__device__ __forceinline__ float siluf(float y) {
  return y * (1.0f / (1.0f + __expf(-y)));
}

constexpr int kB    = 16;
constexpr int kL0   = 12000;
constexpr int kNRow = 16000;   // B * 1000
constexpr int kCB   = 8192;
constexpr int kPad  = 64;      // guard rows per side, per batch (zeroed once)

// padded activation addressing: buf[b][kPad + t][64], batch stride = (L + 2*kPad)
__device__ __forceinline__ size_t prow_off(int b, int stride, int t) {
  return ((size_t)b * stride + kPad + t) * 64;
}

// ---------------- pad zeroing (once per launch; valid-region writes never touch pads) ----
__global__ void zero_pads_kernel(bfr* __restrict__ buf, int L)
{
  const int gid = blockIdx.x * blockDim.x + threadIdx.x;
  if (gid >= kB * 2 * kPad * 64) return;
  const int c = gid & 63;
  const int p = (gid >> 6) & (2 * kPad - 1);
  const int b = gid >> 13;
  const int row = (p < kPad) ? p : (kPad + L + (p - kPad));
  buf[((size_t)b * (L + 2 * kPad) + row) * 64 + c] = 0;
}

// ---------------- weight pre-pack into WMMA A-fragment order ----------------
// out[((j*2+kc)*4 + coGroup)*256 + lane*8 + v8]  (uint = 2 bf16 K-pair)
// transposed=0: enc conv, w layout (cout, cin, KT), tap j
// transposed=1: convT,    w layout (cin, cout, KT), flipped tap KT-1-j
__global__ void pack_w_kernel(const float* __restrict__ w, int KT, int transposed,
                              unsigned int* __restrict__ out)
{
  const int gid = blockIdx.x * blockDim.x + threadIdx.x;
  if (gid >= KT * 2048) return;
  const int v8   = gid & 7;
  const int lane = (gid >> 3) & 31;
  const int cg   = (gid >> 8) & 3;
  const int kc   = (gid >> 10) & 1;
  const int j    = gid >> 11;
  const int cout = cg * 16 + (lane & 15);
  const int hk   = lane >> 4;
  const int kp   = ((v8 >> 2) * 16) + hk * 8 + ((v8 & 3) * 2);
  const int ci   = kc * 32 + kp;
  float w0, w1;
  if (transposed) {
    const int jw = KT - 1 - j;
    w0 = w[((size_t)ci       * 64 + cout) * KT + jw];
    w1 = w[((size_t)(ci + 1) * 64 + cout) * KT + jw];
  } else {
    w0 = w[((size_t)cout * 64 + ci)     * KT + j];
    w1 = w[((size_t)cout * 64 + ci + 1) * KT + j];
  }
  out[gid] = pack_bf2(w0, w1);
}

// ---------------- fold bias + BN into (bias, gain, shift) ----------------
__global__ void prep_ep_kernel(const float* __restrict__ bias,
                               const float* __restrict__ g, const float* __restrict__ b,
                               const float* __restrict__ m, const float* __restrict__ v,
                               float* __restrict__ ep)   // [3][64]
{
  const int c = threadIdx.x;
  const float gain = g[c] * rsqrtf(v[c] + 1e-5f);
  ep[c]        = bias[c];
  ep[64 + c]   = gain;
  ep[128 + c]  = b[c] - m[c] * gain;
}

// ---------------- enc layer 0: Cin=1, k=5, s=1, p=2 (scalar, fused silu+bn) ----------------
__global__ void enc0_kernel(const float* __restrict__ x, const float* __restrict__ w,
                            const float* __restrict__ bias,
                            const float* __restrict__ bng, const float* __restrict__ bnb,
                            const float* __restrict__ bnm, const float* __restrict__ bnv,
                            bfr* __restrict__ out)
{
  const int gid = blockIdx.x * blockDim.x + threadIdx.x;
  if (gid >= kB * kL0) return;
  const int b = gid / kL0, t = gid - b * kL0;
  float xv[5];
#pragma unroll
  for (int j = 0; j < 5; ++j) {
    int ti = t - 2 + j;
    xv[j] = (ti >= 0 && ti < kL0) ? x[(size_t)b * kL0 + ti] : 0.0f;
  }
  bfr* orow = out + prow_off(b, kL0 + 2 * kPad, t);
#pragma unroll
  for (int c0 = 0; c0 < 64; c0 += 8) {
    float rv[8];
#pragma unroll
    for (int i = 0; i < 8; ++i) {
      const int c = c0 + i;
      float acc = bias[c];
#pragma unroll
      for (int j = 0; j < 5; ++j) acc += w[c * 5 + j] * xv[j];
      float s = siluf(acc);
      rv[i] = (s - bnm[c]) * rsqrtf(bnv[c] + 1e-5f) * bng[c] + bnb[c];
    }
    uint4 st;
    st.x = pack_bf2(rv[0], rv[1]); st.y = pack_bf2(rv[2], rv[3]);
    st.z = pack_bf2(rv[4], rv[5]); st.w = pack_bf2(rv[6], rv[7]);
    *(uint4*)(orow + c0) = st;
  }
}

// ---------------- encoder conv (Cin=Cout=64) via WMMA, pre-packed weights ----------------
template<int KT, int ST, int PT, bool WRITE_F32>
__global__ __launch_bounds__(128) void conv_enc_wmma(
    const bfr* __restrict__ in, int inStride,
    const unsigned int* __restrict__ wpk, const float* __restrict__ ep,
    bfr* __restrict__ out, int outStride, int Lout, float* __restrict__ zf32)
{
  const int lane   = threadIdx.x & 31;
  const int wv     = threadIdx.x >> 5;      // 4 waves -> 4 cout groups of 16
  const int b  = blockIdx.y;
  const int t0 = blockIdx.x * 16;
  const int m  = lane & 15;                 // A row (cout) == B col (position)
  const int hk = lane >> 4;

  f32x8 acc = {0.f,0.f,0.f,0.f,0.f,0.f,0.f,0.f};

#pragma unroll
  for (int j = 0; j < KT; ++j) {
    const int tin = (t0 + m) * ST - PT + j;
    const bfr* prow = in + prow_off(b, inStride, tin);   // pads guarantee zeros OOB
#pragma unroll
    for (int kc = 0; kc < 2; ++kc) {
      const unsigned int* aw = wpk + ((((size_t)j * 2 + kc) * 4 + wv) * 32 + lane) * 8;
      FragBF A, Bf;
#pragma unroll
      for (int v8 = 0; v8 < 8; ++v8) {
        A.u[v8]  = aw[v8];
        Bf.u[v8] = *(const unsigned int*)(prow + kc * 32 + (v8 >> 2) * 16 + hk * 8 + (v8 & 3) * 2);
      }
      acc = __builtin_amdgcn_wmma_f32_16x16x32_bf16(false, A.v, false, Bf.v,
                                                    (short)0, acc, false, false);
    }
  }

  const int t = t0 + m;
  if (t < Lout) {
    const int c0 = wv * 16 + hk * 8;
    float rv[8];
#pragma unroll
    for (int i = 0; i < 8; ++i) {
      const int c = c0 + i;
      float y = acc[i] + ep[c];
      rv[i] = siluf(y) * ep[64 + c] + ep[128 + c];
    }
    uint4 st;
    st.x = pack_bf2(rv[0], rv[1]); st.y = pack_bf2(rv[2], rv[3]);
    st.z = pack_bf2(rv[4], rv[5]); st.w = pack_bf2(rv[6], rv[7]);
    *(uint4*)(out + prow_off(b, outStride, t) + c0) = st;
    if (WRITE_F32) {
      float* zp = zf32 + ((size_t)b * Lout + t) * 64 + c0;   // tight layout for loss
      float4 f0 = {rv[0], rv[1], rv[2], rv[3]};
      float4 f1 = {rv[4], rv[5], rv[6], rv[7]};
      *(float4*)zp = f0;  *(float4*)(zp + 4) = f1;
    }
  }
}

// ---------------- transposed conv via phase decomposition + WMMA, pre-packed weights ------
// t = q*ST + r; taps j with j == (KT-1-PT - r) mod ST, qin = q + (r+j-(KT-1-PT))/ST
template<int KT, int ST, int PT>
__global__ __launch_bounds__(128) void convT_wmma(
    const bfr* __restrict__ in, int inStride,
    const unsigned int* __restrict__ wpk, const float* __restrict__ ep,
    bfr* __restrict__ out, int outStride, int Lout)
{
  const int lane = threadIdx.x & 31;
  const int wv   = threadIdx.x >> 5;
  const int b  = blockIdx.z;
  const int r  = blockIdx.y;                // output phase
  const int q0 = blockIdx.x * 16;
  const int m  = lane & 15;
  const int hk = lane >> 4;

  const int KP = KT - 1 - PT;
  const int jr = ((KP - r) % ST + ST) % ST;

  f32x8 acc = {0.f,0.f,0.f,0.f,0.f,0.f,0.f,0.f};

#pragma unroll
  for (int jj = 0; jj < KT / ST; ++jj) {
    const int j  = jr + jj * ST;
    const int dq = (r + j - KP) / ST;       // exact multiple by construction
    const bfr* prow = in + prow_off(b, inStride, q0 + m + dq);
#pragma unroll
    for (int kc = 0; kc < 2; ++kc) {
      const unsigned int* aw = wpk + ((((size_t)j * 2 + kc) * 4 + wv) * 32 + lane) * 8;
      FragBF A, Bf;
#pragma unroll
      for (int v8 = 0; v8 < 8; ++v8) {
        A.u[v8]  = aw[v8];
        Bf.u[v8] = *(const unsigned int*)(prow + kc * 32 + (v8 >> 2) * 16 + hk * 8 + (v8 & 3) * 2);
      }
      acc = __builtin_amdgcn_wmma_f32_16x16x32_bf16(false, A.v, false, Bf.v,
                                                    (short)0, acc, false, false);
    }
  }

  const int t = (q0 + m) * ST + r;
  if (t < Lout) {
    const int c0 = wv * 16 + hk * 8;
    float rv[8];
#pragma unroll
    for (int i = 0; i < 8; ++i) {
      const int c = c0 + i;
      float y = acc[i] + ep[c];
      rv[i] = siluf(y) * ep[64 + c] + ep[128 + c];
    }
    uint4 st;
    st.x = pack_bf2(rv[0], rv[1]); st.y = pack_bf2(rv[2], rv[3]);
    st.z = pack_bf2(rv[4], rv[5]); st.w = pack_bf2(rv[6], rv[7]);
    *(uint4*)(out + prow_off(b, outStride, t) + c0) = st;
  }
}

// ---------------- codebook prep: fp32 -> bf16 + row norms ----------------
__global__ void cb_prep_kernel(const float* __restrict__ cb, bfr* __restrict__ cbb,
                               float* __restrict__ cnorm)
{
  const int code = blockIdx.x;
  const int tid  = threadIdx.x;   // 64 threads
  const float v = cb[(size_t)code * 64 + tid];
  cbb[(size_t)code * 64 + tid] = f2bf(v);
  float s = v * v;
#pragma unroll
  for (int off = 16; off > 0; off >>= 1) s += __shfl_down(s, off, 32);
  __shared__ float sh[2];
  if ((tid & 31) == 0) sh[tid >> 5] = s;
  __syncthreads();
  if (tid == 0) cnorm[code] = sh[0] + sh[1];
}

// ---------------- VQ argmin: 2x16 z-rows x 8192 codes per wave (B-fragment reuse) --------
__global__ __launch_bounds__(128) void vq_argmin_wmma(
    const bfr* __restrict__ z, const bfr* __restrict__ cbb,
    const float* __restrict__ cnorm, int* __restrict__ idxOut)
{
  const int lane = threadIdx.x & 31;
  const int wv   = threadIdx.x >> 5;
  const int row0 = (blockIdx.x * 4 + wv) * 32;   // 500 waves = 125 blocks * 4
  const int m  = lane & 15;
  const int hk = lane >> 4;

  // A fragments for two row tiles (z is padded [b][1128][64]; rows may cross batches)
  FragBF A0a, A1a, A0b, A1b;
  {
    const int ra = row0 + m;       const int ba = ra / 1000, ta = ra - ba * 1000;
    const int rb = row0 + 16 + m;  const int bb = rb / 1000, tb = rb - bb * 1000;
    const bfr* za = z + prow_off(ba, 1000 + 2 * kPad, ta);
    const bfr* zb = z + prow_off(bb, 1000 + 2 * kPad, tb);
#pragma unroll
    for (int v8 = 0; v8 < 8; ++v8) {
      const int kp = ((v8 >> 2) * 16) + hk * 8 + ((v8 & 3) * 2);
      A0a.u[v8] = *(const unsigned int*)(za + kp);
      A1a.u[v8] = *(const unsigned int*)(za + 32 + kp);
      A0b.u[v8] = *(const unsigned int*)(zb + kp);
      A1b.u[v8] = *(const unsigned int*)(zb + 32 + kp);
    }
  }

  float bestA[8], bestB[8]; int idxA[8], idxB[8];
#pragma unroll
  for (int i = 0; i < 8; ++i) { bestA[i] = 3.4e38f; bestB[i] = 3.4e38f; idxA[i] = 0; idxB[i] = 0; }

  for (int c0 = 0; c0 < kCB; c0 += 16) {
    const int code = c0 + m;
    const bfr* crow = cbb + (size_t)code * 64;
    __builtin_prefetch(crow + 16 * 64, 0, 1);   // next code tile
    FragBF B0, B1;
#pragma unroll
    for (int v8 = 0; v8 < 8; ++v8) {
      const int ci = ((v8 >> 2) * 16) + hk * 8 + ((v8 & 3) * 2);
      B0.u[v8] = *(const unsigned int*)(crow + ci);
      B1.u[v8] = *(const unsigned int*)(crow + 32 + ci);
    }
    f32x8 accA = {0.f,0.f,0.f,0.f,0.f,0.f,0.f,0.f};
    f32x8 accB = {0.f,0.f,0.f,0.f,0.f,0.f,0.f,0.f};
    accA = __builtin_amdgcn_wmma_f32_16x16x32_bf16(false, A0a.v, false, B0.v, (short)0, accA, false, false);
    accB = __builtin_amdgcn_wmma_f32_16x16x32_bf16(false, A0b.v, false, B0.v, (short)0, accB, false, false);
    accA = __builtin_amdgcn_wmma_f32_16x16x32_bf16(false, A1a.v, false, B1.v, (short)0, accA, false, false);
    accB = __builtin_amdgcn_wmma_f32_16x16x32_bf16(false, A1b.v, false, B1.v, (short)0, accB, false, false);
    const float cn = cnorm[code];
#pragma unroll
    for (int i = 0; i < 8; ++i) {
      float ca = cn - 2.0f * accA[i];           // |z|^2 constant per row -> dropped
      float cb_ = cn - 2.0f * accB[i];
      if (ca  < bestA[i]) { bestA[i] = ca;  idxA[i] = code; }
      if (cb_ < bestB[i]) { bestB[i] = cb_; idxB[i] = code; }
    }
  }

  // reduce over 16 lanes of each half (rows 0-7 in lanes 0-15, rows 8-15 in lanes 16-31)
#pragma unroll
  for (int i = 0; i < 8; ++i) {
#pragma unroll
    for (int off = 1; off < 16; off <<= 1) {
      float oa = __shfl_xor(bestA[i], off, 32);
      int   ia = __shfl_xor(idxA[i],  off, 32);
      if (oa < bestA[i] || (oa == bestA[i] && ia < idxA[i])) { bestA[i] = oa; idxA[i] = ia; }
      float ob = __shfl_xor(bestB[i], off, 32);
      int   ib = __shfl_xor(idxB[i],  off, 32);
      if (ob < bestB[i] || (ob == bestB[i] && ib < idxB[i])) { bestB[i] = ob; idxB[i] = ib; }
    }
  }
  if (m == 0) {
#pragma unroll
    for (int i = 0; i < 8; ++i) {
      idxOut[row0 + hk * 8 + i]      = idxA[i];
      idxOut[row0 + 16 + hk * 8 + i] = idxB[i];
    }
  }
}

// ---------------- gather quant rows, per-row commit-loss partials, idx -> float ----------
__global__ void vq_gather_kernel(const int* __restrict__ idx, const float* __restrict__ cb,
                                 const float* __restrict__ zf, bfr* __restrict__ quant,
                                 float* __restrict__ outIdx, float* __restrict__ rowLoss)
{
  const int row = blockIdx.x;
  const int tid = threadIdx.x;   // 64
  const int c   = idx[row];
  const int b = row / 1000, t = row - b * 1000;
  const float q  = cb[(size_t)c * 64 + tid];
  const float zv = zf[(size_t)row * 64 + tid];
  quant[prow_off(b, 1000 + 2 * kPad, t) + tid] = f2bf(q);
  float d = q - zv; d *= d;
#pragma unroll
  for (int off = 16; off > 0; off >>= 1) d += __shfl_down(d, off, 32);
  __shared__ float sh[2];
  if ((tid & 31) == 0) sh[tid >> 5] = d;
  __syncthreads();
  if (tid == 0) { rowLoss[row] = sh[0] + sh[1]; outIdx[row] = (float)c; }
}

__global__ void loss_finalize_kernel(const float* __restrict__ rowLoss, float* __restrict__ outLoss)
{
  __shared__ float sh[256];
  const int tid = threadIdx.x;
  float s = 0.f;
  for (int i = tid; i < kNRow; i += 256) s += rowLoss[i];  // fixed order -> deterministic
  sh[tid] = s;
  __syncthreads();
  for (int off = 128; off > 0; off >>= 1) { if (tid < off) sh[tid] += sh[tid + off]; __syncthreads(); }
  if (tid == 0) *outLoss = 2.0f * sh[0] / (float)(kNRow * 64);
}

// ---------------- head: 1x1 conv 64->1, pad last sample to L=12000 ----------------
__global__ void head_kernel(const bfr* __restrict__ h, int hStride,
                            const float* __restrict__ w,
                            const float* __restrict__ bias, float* __restrict__ out)
{
  const int gid = blockIdx.x * blockDim.x + threadIdx.x;
  if (gid >= kB * kL0) return;
  const int b = gid / kL0, t = gid - b * kL0;
  float r = 0.f;
  if (t < 11999) {                       // decoder output length = 11999, tail padded 0
    const bfr* p = h + prow_off(b, hStride, t);
    float acc = bias[0];
#pragma unroll
    for (int c = 0; c < 64; ++c) acc += bf2f(p[c]) * w[c];
    r = acc;
  }
  out[gid] = r;
}

// ---------------- launcher ----------------
extern "C" void kernel_launch(void* const* d_in, const int* in_sizes, int n_in,
                              void* d_out, int out_size, void* d_ws, size_t ws_size,
                              hipStream_t stream)
{
  auto f = [&](int i) -> const float* { return (const float*)d_in[i]; };
  struct LayerP { const float *bnb,*bng,*bnm,*bnv,*cbias,*cw; };
  LayerP enc[4], dec[3];
  const float *p_cb, *p_x, *p_hb, *p_hw;

  if (in_sizes[0] == kCB * 64) {
    // jax tree (sorted-key) flattening: codebook, dec[0..2]{bn.b,g,m,v, conv.b,w},
    // enc[0..3]{...}, head{b,w}, x
    p_cb = f(0);
    int k = 1;
    for (int l = 0; l < 3; ++l) { dec[l] = { f(k), f(k+1), f(k+2), f(k+3), f(k+4), f(k+5) }; k += 6; }
    for (int l = 0; l < 4; ++l) { enc[l] = { f(k), f(k+1), f(k+2), f(k+3), f(k+4), f(k+5) }; k += 6; }
    p_hb = f(43); p_hw = f(44); p_x = f(45);
  } else {
    // insertion-order flattening: x, enc[0..3]{w,b, g,b,m,v}, dec[0..2]{...}, head{w,b}, cb
    p_x = f(0);
    int k = 1;
    for (int l = 0; l < 4; ++l) { enc[l].cw = f(k); enc[l].cbias = f(k+1);
      enc[l].bng = f(k+2); enc[l].bnb = f(k+3); enc[l].bnm = f(k+4); enc[l].bnv = f(k+5); k += 6; }
    for (int l = 0; l < 3; ++l) { dec[l].cw = f(k); dec[l].cbias = f(k+1);
      dec[l].bng = f(k+2); dec[l].bnb = f(k+3); dec[l].bnm = f(k+4); dec[l].bnv = f(k+5); k += 6; }
    p_hw = f(43); p_hb = f(44); p_cb = f(45);
  }
  (void)n_in; (void)out_size;

  // workspace carve
  char* ws = (char*)d_ws; size_t off = 0;
  auto carve = [&](size_t bytes) -> void* {
    void* p = ws + off; off = (off + bytes + 255) & ~(size_t)255; return p;
  };
  const int PAD2 = 2 * kPad;
  bfr*   actA = (bfr*)  carve((size_t)kB * (12000 + PAD2) * 64 * 2);  // enc0 out / dec3 out
  bfr*   actB = (bfr*)  carve((size_t)kB * (4000  + PAD2) * 64 * 2);  // enc1 out / dec2 out
  bfr*   actC = (bfr*)  carve((size_t)kB * (2000  + PAD2) * 64 * 2);  // enc2 out / dec1 out
  bfr*   zbf  = (bfr*)  carve((size_t)kB * (1000  + PAD2) * 64 * 2);
  bfr*   qnt  = (bfr*)  carve((size_t)kB * (1000  + PAD2) * 64 * 2);
  float* zf   = (float*)carve((size_t)kNRow * 64 * 4);
  bfr*   cbb  = (bfr*)  carve((size_t)kCB * 64 * 2);
  float* cn   = (float*)carve((size_t)kCB * 4);
  int*   idxb = (int*)  carve((size_t)kNRow * 4);
  float* rls  = (float*)carve((size_t)kNRow * 4);
  unsigned int* wpkE1 = (unsigned int*)carve((size_t)9  * 2048 * 4);
  unsigned int* wpkE2 = (unsigned int*)carve((size_t)9  * 2048 * 4);
  unsigned int* wpkE3 = (unsigned int*)carve((size_t)5  * 2048 * 4);
  unsigned int* wpkD1 = (unsigned int*)carve((size_t)8  * 2048 * 4);
  unsigned int* wpkD2 = (unsigned int*)carve((size_t)12 * 2048 * 4);
  unsigned int* wpkD3 = (unsigned int*)carve((size_t)18 * 2048 * 4);
  float* epE1 = (float*)carve(192 * 4);
  float* epE2 = (float*)carve(192 * 4);
  float* epE3 = (float*)carve(192 * 4);
  float* epD1 = (float*)carve(192 * 4);
  float* epD2 = (float*)carve(192 * 4);
  float* epD3 = (float*)carve(192 * 4);
  (void)ws_size;

  float* recon   = (float*)d_out;            // 16*12000
  float* outIdx  = recon + kB * kL0;         // 16000 (argmin indices as float)
  float* outLoss = outIdx + kNRow;           // 1

  // ---- prep: pads, packed weights, folded epilogues, codebook ----
  const int padThreads = kB * 2 * kPad * 64;
  zero_pads_kernel<<<(padThreads + 255) / 256, 256, 0, stream>>>(actA, 12000);
  zero_pads_kernel<<<(padThreads + 255) / 256, 256, 0, stream>>>(actB, 4000);
  zero_pads_kernel<<<(padThreads + 255) / 256, 256, 0, stream>>>(actC, 2000);
  zero_pads_kernel<<<(padThreads + 255) / 256, 256, 0, stream>>>(qnt, 1000);

  pack_w_kernel<<<(9  * 2048 + 255) / 256, 256, 0, stream>>>(enc[1].cw, 9,  0, wpkE1);
  pack_w_kernel<<<(9  * 2048 + 255) / 256, 256, 0, stream>>>(enc[2].cw, 9,  0, wpkE2);
  pack_w_kernel<<<(5  * 2048 + 255) / 256, 256, 0, stream>>>(enc[3].cw, 5,  0, wpkE3);
  pack_w_kernel<<<(8  * 2048 + 255) / 256, 256, 0, stream>>>(dec[0].cw, 8,  1, wpkD1);
  pack_w_kernel<<<(12 * 2048 + 255) / 256, 256, 0, stream>>>(dec[1].cw, 12, 1, wpkD2);
  pack_w_kernel<<<(18 * 2048 + 255) / 256, 256, 0, stream>>>(dec[2].cw, 18, 1, wpkD3);

  prep_ep_kernel<<<1, 64, 0, stream>>>(enc[1].cbias, enc[1].bng, enc[1].bnb, enc[1].bnm, enc[1].bnv, epE1);
  prep_ep_kernel<<<1, 64, 0, stream>>>(enc[2].cbias, enc[2].bng, enc[2].bnb, enc[2].bnm, enc[2].bnv, epE2);
  prep_ep_kernel<<<1, 64, 0, stream>>>(enc[3].cbias, enc[3].bng, enc[3].bnb, enc[3].bnm, enc[3].bnv, epE3);
  prep_ep_kernel<<<1, 64, 0, stream>>>(dec[0].cbias, dec[0].bng, dec[0].bnb, dec[0].bnm, dec[0].bnv, epD1);
  prep_ep_kernel<<<1, 64, 0, stream>>>(dec[1].cbias, dec[1].bng, dec[1].bnb, dec[1].bnm, dec[1].bnv, epD2);
  prep_ep_kernel<<<1, 64, 0, stream>>>(dec[2].cbias, dec[2].bng, dec[2].bnb, dec[2].bnm, dec[2].bnv, epD3);

  cb_prep_kernel<<<kCB, 64, 0, stream>>>(p_cb, cbb, cn);

  // ---- encoder ----
  enc0_kernel<<<(kB * kL0 + 255) / 256, 256, 0, stream>>>(
      p_x, enc[0].cw, enc[0].cbias, enc[0].bng, enc[0].bnb, enc[0].bnm, enc[0].bnv, actA);
  conv_enc_wmma<9, 3, 4, false><<<dim3(250, kB), 128, 0, stream>>>(
      actA, 12000 + PAD2, wpkE1, epE1, actB, 4000 + PAD2, 4000, nullptr);
  conv_enc_wmma<9, 2, 4, false><<<dim3(125, kB), 128, 0, stream>>>(
      actB, 4000 + PAD2, wpkE2, epE2, actC, 2000 + PAD2, 2000, nullptr);
  conv_enc_wmma<5, 2, 2, true><<<dim3(63, kB), 128, 0, stream>>>(
      actC, 2000 + PAD2, wpkE3, epE3, zbf, 1000 + PAD2, 1000, zf);

  // ---- VQ ----
  vq_argmin_wmma<<<125, 128, 0, stream>>>(zbf, cbb, cn, idxb);
  vq_gather_kernel<<<kNRow, 64, 0, stream>>>(idxb, p_cb, zf, qnt, outIdx, rls);

  // ---- decoder (phase-decomposed transposed convs) ----
  convT_wmma<8, 2, 3><<<dim3(63, 2, kB), 128, 0, stream>>>(
      qnt, 1000 + PAD2, wpkD1, epD1, actC, 2000 + PAD2, 2000);
  convT_wmma<12, 2, 5><<<dim3(125, 2, kB), 128, 0, stream>>>(
      actC, 2000 + PAD2, wpkD2, epD2, actB, 4000 + PAD2, 4000);
  convT_wmma<18, 3, 8><<<dim3(250, 3, kB), 128, 0, stream>>>(
      actB, 4000 + PAD2, wpkD3, epD3, actA, 11999 + PAD2, 11999);

  // ---- head + outputs ----
  head_kernel<<<(kB * kL0 + 255) / 256, 256, 0, stream>>>(actA, 11999 + PAD2, p_hw, p_hb, recon);
  loss_finalize_kernel<<<1, 256, 0, stream>>>(rls, outLoss);
}
